// ROI_ALIGN_29703993819269
// MI455X (gfx1250) — compile-verified, matrix operand order checked by reading, not run on hardware
//
#include <hip/hip_runtime.h>

// ---------------------------------------------------------------------------
// MI455X / gfx1250, wave32. GEMMs run on V_WMMA_F32_16X16X32_F16 (f16 in,
// fp32 accumulate). Problem sizes: M=6272, K1=2048,N1=512, K2=512,N2=256.
//
// Pipeline:
//   0) weights -> f16, BN folded into per-channel scale/bias
//   1) NCHW fp32 -> NHWC f16 transpose (LDS-tiled, coalesced both sides)
//      [skipped if workspace too small; ROI falls back to NCHW gathers]
//   2) ROI-Align -> x[6272][2048] f16 (channel-contiguous taps when NHWC)
//   3) WMMA GEMM1 + BN + ReLU -> h1[6272][512] f16
//   4) WMMA GEMM2 + BN + ReLU -> h2[6272][256] f16
//   5) mean over 49 bins -> out[128][256][1][1] fp32
// ---------------------------------------------------------------------------

typedef __attribute__((ext_vector_type(16))) _Float16 v16h;
typedef __attribute__((ext_vector_type(8)))  _Float16 v8h;
typedef __attribute__((ext_vector_type(8)))  float    v8f;

union V16 { v16h v; v8h h[2]; };

#define NBOX  128
#define KBIN  7
#define NBINS 49            // 7*7
#define MROWS (NBOX*NBINS)  // 6272 = 392 * 16
#define C0    2048
#define C1    512
#define C2    256
#define FH    64
#define FW    64
#define FHW   (FH*FW)       // 4096
#define EPSBN 1e-5f

// ---------------------------------------------------------------------------
// fp32 -> f16 weight conversion (one-shot, tiny)
// ---------------------------------------------------------------------------
__global__ void cvt_f16_kernel(const float* __restrict__ in,
                               _Float16* __restrict__ out, int n) {
  int i = blockIdx.x * blockDim.x + threadIdx.x;
  if (i < n) out[i] = (_Float16)in[i];
}

// ---------------------------------------------------------------------------
// Fold conv-bias + BN into per-output-channel scale/bias:
//   relu((conv + b)*s + (be - m*s)) with s = g/sqrt(v+eps)
//   => relu(conv*s + (b*s + be - m*s))
// ---------------------------------------------------------------------------
__global__ void fold_bn_kernel(const float* __restrict__ b1, const float* __restrict__ g1,
                               const float* __restrict__ be1, const float* __restrict__ m1,
                               const float* __restrict__ v1,
                               const float* __restrict__ b2, const float* __restrict__ g2,
                               const float* __restrict__ be2, const float* __restrict__ m2,
                               const float* __restrict__ v2,
                               float* __restrict__ s1, float* __restrict__ t1,
                               float* __restrict__ s2, float* __restrict__ t2) {
  int i = blockIdx.x * blockDim.x + threadIdx.x;
  if (i < C1) {
    float s = g1[i] * rsqrtf(v1[i] + EPSBN);
    s1[i] = s;
    t1[i] = b1[i] * s + be1[i] - m1[i] * s;
  }
  if (i < C2) {
    float s = g2[i] * rsqrtf(v2[i] + EPSBN);
    s2[i] = s;
    t2[i] = b2[i] * s + be2[i] - m2[i] * s;
  }
}

// ---------------------------------------------------------------------------
// NCHW fp32 -> NHWC f16, 64x64 tiles via LDS (coalesced reads AND writes).
//   grid = (FHW/64, C0/64, B), block = 256
// ---------------------------------------------------------------------------
__global__ void __launch_bounds__(256)
nchw_to_nhwc_f16_kernel(const float* __restrict__ feat,
                        _Float16* __restrict__ nhwc) {
  __shared__ float tile[64][65];  // +1 pad: no bank conflicts on transpose
  const int b   = blockIdx.z;
  const int hw0 = blockIdx.x * 64;
  const int c0  = blockIdx.y * 64;
  const int tx  = threadIdx.x & 63;   // fast index
  const int ty  = threadIdx.x >> 6;   // 0..3

  // read: contiguous along hw (input fast dim)
#pragma unroll
  for (int cc = ty; cc < 64; cc += 4)
    tile[cc][tx] = feat[((size_t)b * C0 + (c0 + cc)) * FHW + hw0 + tx];
  __syncthreads();

  // write: contiguous along c (output fast dim)
#pragma unroll
  for (int ww = ty; ww < 64; ww += 4)
    nhwc[((size_t)b * FHW + (hw0 + ww)) * C0 + c0 + tx] =
        (_Float16)tile[tx][ww];
}

// ---------------------------------------------------------------------------
// Shared ROI tap computation: 16 (offset, weight) pairs, uniform over channels
// ---------------------------------------------------------------------------
__device__ __forceinline__ void roi_taps(const float* __restrict__ boxes,
                                         int n, int ph, int pw,
                                         int& b, float* wgt, int* off) {
  const float* box = boxes + n * 5;
  b = (int)box[0];
  const float x1 = box[1], y1 = box[2], x2 = box[3], y2 = box[4];
  const float bw = fmaxf(x2 - x1, 1.0f) * (1.0f / KBIN);
  const float bh = fmaxf(y2 - y1, 1.0f) * (1.0f / KBIN);
#pragma unroll
  for (int iy = 0; iy < 2; ++iy) {
    const float y  = y1 + ((float)ph + (iy + 0.5f) * 0.5f) * bh;
    const bool  vy = (y >= -1.0f) && (y <= (float)FH);
    const float yc = fminf(fmaxf(y, 0.0f), (float)(FH - 1));
    const int   y0 = (int)floorf(yc);
    const int   yh = min(y0 + 1, FH - 1);
    const float ly = yc - (float)y0;
#pragma unroll
    for (int ix = 0; ix < 2; ++ix) {
      const float x  = x1 + ((float)pw + (ix + 0.5f) * 0.5f) * bw;
      const bool  vx = (x >= -1.0f) && (x <= (float)FW);
      const float xc = fminf(fmaxf(x, 0.0f), (float)(FW - 1));
      const int   x0 = (int)floorf(xc);
      const int   xh = min(x0 + 1, FW - 1);
      const float lx = xc - (float)x0;
      const float vm = (vy && vx) ? 1.0f : 0.0f;
      const int s = (iy * 2 + ix) * 4;
      wgt[s + 0] = vm * (1.0f - ly) * (1.0f - lx); off[s + 0] = y0 * FW + x0;
      wgt[s + 1] = vm * (1.0f - ly) * lx;          off[s + 1] = y0 * FW + xh;
      wgt[s + 2] = vm * ly * (1.0f - lx);          off[s + 2] = yh * FW + x0;
      wgt[s + 3] = vm * ly * lx;                   off[s + 3] = yh * FW + xh;
    }
  }
}

// ---------------------------------------------------------------------------
// ROI-Align from NHWC f16: each tap is a contiguous 2048-ch f16 vector ->
// fully coalesced gathers out of L2. One block per (box, bin).
// ---------------------------------------------------------------------------
__global__ void __launch_bounds__(256)
roi_align_nhwc_f16_kernel(const _Float16* __restrict__ nhwc,
                          const float* __restrict__ boxes,
                          _Float16* __restrict__ xout) {
  const int bin = blockIdx.x % NBINS;
  const int n   = blockIdx.x / NBINS;
  float wgt[16]; int off[16]; int b;
  roi_taps(boxes, n, bin / KBIN, bin % KBIN, b, wgt, off);

  const _Float16* fb   = nhwc + (size_t)b * FHW * C0;
  _Float16*       orow = xout + (size_t)blockIdx.x * C0;

  for (int c = threadIdx.x; c < C0; c += blockDim.x) {
    float acc = 0.0f;
#pragma unroll
    for (int j = 0; j < 16; ++j)
      acc += wgt[j] * (float)fb[(size_t)off[j] * C0 + c];
    orow[c] = (_Float16)(acc * 0.25f);  // mean over 2x2 samples
  }
}

// ---------------------------------------------------------------------------
// ROI-Align fallback from NCHW fp32 (workspace too small for NHWC staging)
// ---------------------------------------------------------------------------
__global__ void __launch_bounds__(256)
roi_align_nchw_f16_kernel(const float* __restrict__ feat,
                          const float* __restrict__ boxes,
                          _Float16* __restrict__ xout) {
  const int bin = blockIdx.x % NBINS;
  const int n   = blockIdx.x / NBINS;
  float wgt[16]; int off[16]; int b;
  roi_taps(boxes, n, bin / KBIN, bin % KBIN, b, wgt, off);

  const float* fb   = feat + (size_t)b * C0 * FHW;
  _Float16*    orow = xout + (size_t)blockIdx.x * C0;

  for (int c = threadIdx.x; c < C0; c += blockDim.x) {
    const float* fc = fb + (size_t)c * FHW;
    float acc = 0.0f;
#pragma unroll
    for (int j = 0; j < 16; ++j) acc += wgt[j] * fc[off[j]];
    orow[c] = (_Float16)(acc * 0.25f);
  }
}

// ---------------------------------------------------------------------------
// Fused GEMM + per-channel scale/bias + ReLU, f16 in / f16 out, fp32 accum.
//   A: [M][K] row-major f16 (activations)
//   B: [N][K] row-major f16 (weights, (O,C) layout -> K contiguous per col)
// Block = 128 threads = 4 waves; each wave computes a 16(M) x 64(N) tile:
// one A fragment reused across four B fragments -> 4 WMMAs per 32-wide K step.
// Fragment addressing follows the ISA 16-bit A-matrix layout:
//   lanes 0-15 hold M=lane, K = {0..7, 16..23}; lanes 16-31 K = {8..15, 24..31}
// ---------------------------------------------------------------------------
__global__ void __launch_bounds__(128)
gemm_bn_relu_f16_kernel(const _Float16* __restrict__ A,
                        const _Float16* __restrict__ B,
                        const float* __restrict__ scale,
                        const float* __restrict__ bias,
                        _Float16* __restrict__ Out,
                        int M, int K, int N) {
  const int lane  = threadIdx.x & 31;
  const int wave  = threadIdx.x >> 5;
  const int ksel  = lane >> 4;     // which K half-group this lane holds
  const int l16   = lane & 15;     // M (for A) / N (for B) within the tile
  const int mbase = (blockIdx.x * 4 + wave) * 16;
  const int nbase = blockIdx.y * 64;
  if (mbase >= M) return;

  const _Float16* arow = A + (size_t)(mbase + l16) * K + ksel * 8;

  v8f acc[4] = {};

  for (int k0 = 0; k0 < K; k0 += 32) {
    V16 a;
    const v8h* ap = (const v8h*)(arow + k0);
    a.h[0] = ap[0];          // K = k0 + {ksel*8 .. ksel*8+7}
    a.h[1] = ap[2];          // K = k0 + 16 + {ksel*8 .. ksel*8+7}
    // near-cache (WGP-scope) prefetch of the A row two k-steps ahead:
    if (k0 + 32 < K) __builtin_prefetch(arow + k0 + 64, 0, 3);

#pragma unroll
    for (int t = 0; t < 4; ++t) {
      V16 bf;
      const v8h* bp =
          (const v8h*)(B + (size_t)(nbase + t * 16 + l16) * K + k0 + ksel * 8);
      bf.h[0] = bp[0];
      bf.h[1] = bp[2];
      acc[t] = __builtin_amdgcn_wmma_f32_16x16x32_f16(
          /*neg_a=*/false, a.v, /*neg_b=*/false, bf.v,
          /*c_mod=*/(short)0, acc[t], /*reuse_a=*/false, /*reuse_b=*/false);
    }
  }

  // Epilogue: C/D 16x16 f32 layout = 8 VGPRs; VGPR r, lane l:
  //   row = r + (l<16 ? 0 : 8), col = l & 15.
#pragma unroll
  for (int t = 0; t < 4; ++t) {
    const int col = nbase + t * 16 + l16;
    const float s  = scale[col];
    const float tb = bias[col];
#pragma unroll
    for (int r = 0; r < 8; ++r) {
      const int row = mbase + r + ksel * 8;
      const float v = acc[t][r] * s + tb;
      Out[(size_t)row * N + col] = (_Float16)fmaxf(v, 0.0f);
    }
  }
}

// ---------------------------------------------------------------------------
// Mean over the 49 bins of each box: h2[6272][256] -> out[128][256][1][1]
// (deterministic serial sum, no atomics -> safe for graph replay)
// ---------------------------------------------------------------------------
__global__ void mean_pool_kernel(const _Float16* __restrict__ h2,
                                 float* __restrict__ out) {
  const int i = blockIdx.x * blockDim.x + threadIdx.x;
  if (i >= NBOX * C2) return;
  const int n = i / C2;
  const int o = i % C2;
  const _Float16* p = h2 + (size_t)(n * NBINS) * C2 + o;
  float s = 0.0f;
#pragma unroll
  for (int pb = 0; pb < NBINS; ++pb) s += (float)p[(size_t)pb * C2];
  out[i] = s * (1.0f / (float)NBINS);
}

// ---------------------------------------------------------------------------
extern "C" void kernel_launch(void* const* d_in, const int* in_sizes, int n_in,
                              void* d_out, int out_size, void* d_ws, size_t ws_size,
                              hipStream_t stream) {
  const float* feat  = (const float*)d_in[0];
  const float* boxes = (const float*)d_in[1];
  const float* w1    = (const float*)d_in[2];
  const float* b1    = (const float*)d_in[3];
  const float* g1    = (const float*)d_in[4];
  const float* be1   = (const float*)d_in[5];
  const float* m1    = (const float*)d_in[6];
  const float* v1    = (const float*)d_in[7];
  const float* w2    = (const float*)d_in[8];
  const float* b2    = (const float*)d_in[9];
  const float* g2    = (const float*)d_in[10];
  const float* be2   = (const float*)d_in[11];
  const float* m2    = (const float*)d_in[12];
  const float* v2    = (const float*)d_in[13];
  float* out = (float*)d_out;
  (void)in_sizes; (void)n_in; (void)out_size;

  char* ws = (char*)d_ws;
  size_t off = 0;
  _Float16* xbuf = (_Float16*)(ws + off); off += (size_t)MROWS * C0 * 2;  // 25.7 MB
  _Float16* h1   = (_Float16*)(ws + off); off += (size_t)MROWS * C1 * 2;  //  6.4 MB
  _Float16* h2b  = (_Float16*)(ws + off); off += (size_t)MROWS * C2 * 2;  //  3.2 MB
  _Float16* w1h  = (_Float16*)(ws + off); off += (size_t)C1 * C0 * 2;     //  2.1 MB
  _Float16* w2h  = (_Float16*)(ws + off); off += (size_t)C2 * C1 * 2;     //  0.26 MB
  float* s1 = (float*)(ws + off); off += C1 * sizeof(float);
  float* t1 = (float*)(ws + off); off += C1 * sizeof(float);
  float* s2 = (float*)(ws + off); off += C2 * sizeof(float);
  float* t2 = (float*)(ws + off); off += C2 * sizeof(float);
  off = (off + 255) & ~(size_t)255;
  _Float16* nhwc = (_Float16*)(ws + off);
  const size_t nhwc_bytes = (size_t)4 * FHW * C0 * 2;                     // 67 MB
  const bool use_nhwc = (ws_size >= off + nhwc_bytes);

  // 1) one-shot prep: weights -> f16, BN folding
  cvt_f16_kernel<<<(C1 * C0 + 255) / 256, 256, 0, stream>>>(w1, w1h, C1 * C0);
  cvt_f16_kernel<<<(C2 * C1 + 255) / 256, 256, 0, stream>>>(w2, w2h, C2 * C1);
  fold_bn_kernel<<<2, 256, 0, stream>>>(b1, g1, be1, m1, v1,
                                        b2, g2, be2, m2, v2,
                                        s1, t1, s2, t2);

  // 2) ROI-Align -> x[6272][2048] f16
  if (use_nhwc) {
    nchw_to_nhwc_f16_kernel<<<dim3(FHW / 64, C0 / 64, 4), 256, 0, stream>>>(
        feat, nhwc);
    roi_align_nhwc_f16_kernel<<<MROWS, 256, 0, stream>>>(nhwc, boxes, xbuf);
  } else {
    roi_align_nchw_f16_kernel<<<MROWS, 256, 0, stream>>>(feat, boxes, xbuf);
  }

  // 3) GEMM1 (6272x2048 @ 2048x512) + BN + ReLU -> h1
  gemm_bn_relu_f16_kernel<<<dim3(MROWS / 64, C1 / 64), 128, 0, stream>>>(
      xbuf, w1h, s1, t1, h1, MROWS, C0, C1);

  // 4) GEMM2 (6272x512 @ 512x256) + BN + ReLU -> h2
  gemm_bn_relu_f16_kernel<<<dim3(MROWS / 64, C2 / 64), 128, 0, stream>>>(
      h1, w2h, s2, t2, h2b, MROWS, C1, C2);

  // 5) mean over bins -> out[128*256]
  mean_pool_kernel<<<(NBOX * C2 + 255) / 256, 256, 0, stream>>>(h2b, out);
}
